// MultiHeadAttention_46746424049959
// MI455X (gfx1250) — matrix operand order repaired and checked by reading, not compiled, not run on hardware
//
#include <hip/hip_runtime.h>
#include <hip/hip_bf16.h>

// ---------------------------------------------------------------------------
// MI455X (gfx1250) multi-head attention, f16 WMMA pipeline with f32 accum.
//   b=8, p=1024, d_model=1024, h=16, d_head=64
// Compute-bound (~103 GFLOP vs ~150MB @ 23.3TB/s): everything routed through
// v_wmma_f32_16x16x32_f16. Pure-copy LDS stagings use the CDNA5 async
// global->LDS path (ASYNCcnt) when the toolchain exposes the builtins.
// ---------------------------------------------------------------------------

typedef _Float16 f16;
typedef __attribute__((ext_vector_type(16))) _Float16 v16h;
typedef __attribute__((ext_vector_type(8)))  _Float16 v8h;
typedef __attribute__((ext_vector_type(8)))  float    v8f;
typedef __attribute__((ext_vector_type(4)))  int      v4i;

#define D_MODEL 1024
#define NUM_HEAD 16
#define D_HEAD 64
#define BATCH 8
#define SEQ 1024
#define ROWS (BATCH * SEQ)          // 8192 GEMM rows

// log2(e) / sqrt(64)
#define SOFTMAX_C1 0.18033688011112042f

#if defined(__AMDGCN__) && __has_builtin(__builtin_amdgcn_global_load_async_to_lds_b128)
#define USE_ASYNC_LDS 1
#else
#define USE_ASYNC_LDS 0
#endif

#if USE_ASYNC_LDS
typedef __attribute__((address_space(1))) v4i* gvec_p;   // global B128 payload
typedef __attribute__((address_space(3))) v4i* lvec_p;   // LDS B128 payload

static __device__ __forceinline__ void async_copy_b128(const f16* g, f16* l) {
  __builtin_amdgcn_global_load_async_to_lds_b128((gvec_p)g, (lvec_p)l, 0, 0);
}
static __device__ __forceinline__ void async_drain() {
#if __has_builtin(__builtin_amdgcn_s_wait_asynccnt)
  __builtin_amdgcn_s_wait_asynccnt(0);
#else
  asm volatile("s_wait_asynccnt 0x0" ::: "memory");
#endif
}
#endif

static __device__ __forceinline__ v16h make16(v8h lo, v8h hi) {
  return __builtin_shufflevector(lo, hi, 0,1,2,3,4,5,6,7,8,9,10,11,12,13,14,15);
}

// A fragment (16x32, M x K) for v_wmma_f32_16x16x32_f16.
// rowptr = &src[row_m * ld + k0]; per ISA layout lane(half=lane/16) holds
// K = {half*8 .. half*8+7} and {16+half*8 .. 16+half*8+7}.
static __device__ __forceinline__ v16h load_a_frag(const f16* rowptr, int half) {
  v8h lo = *(const v8h*)(rowptr + half * 8);
  v8h hi = *(const v8h*)(rowptr + 16 + half * 8);
  return make16(lo, hi);
}

// B fragment (32x16, K x N) loaded from an N-major (B-transposed) buffer:
// rowptr = &Bt[n * ld + k0]; lane(half) holds K = {half*8 .. half*8+7} and
// {16+half*8 .. 16+half*8+7}.
static __device__ __forceinline__ v16h load_bt_frag(const f16* rowptr, int half) {
  v8h lo = *(const v8h*)(rowptr + half * 8);
  v8h hi = *(const v8h*)(rowptr + 16 + half * 8);
  return make16(lo, hi);
}

static __device__ __forceinline__ v8f wmma_f16(v16h a, v16h b, v8f c) {
  return __builtin_amdgcn_wmma_f32_16x16x32_f16(false, a, false, b, (short)0, c,
                                                false, false);
}

// ---------------------------------------------------------------------------
// Kernel 1: QKV projection.  Y = x @ W + bias  (M=8192, N=1024, K=1024)
// blockIdx.z selects {w_q, w_k, w_v}; result stored as f16 in [b,h,p,d].
// Block tile 128x128, BK=32, 256 threads = 8 waves in a 2x4 wave grid.
// (Both stagings convert f32->f16 / transpose, so they stay on the VGPR path.)
// ---------------------------------------------------------------------------
__global__ __launch_bounds__(256)
void mha_qkv_proj(const float* __restrict__ x,
                  const float* __restrict__ w_q, const float* __restrict__ b_q,
                  const float* __restrict__ w_k, const float* __restrict__ b_k,
                  const float* __restrict__ w_v, const float* __restrict__ b_v,
                  f16* __restrict__ qbuf, f16* __restrict__ kbuf,
                  f16* __restrict__ vbuf) {
  const int z = blockIdx.z;
  const float* wp = (z == 0) ? w_q : (z == 1) ? w_k : w_v;
  const float* bp = (z == 0) ? b_q : (z == 1) ? b_k : b_v;
  f16* dst = (z == 0) ? qbuf : (z == 1) ? kbuf : vbuf;

  const int m0 = blockIdx.x * 128;
  const int n0 = blockIdx.y * 128;
  const int tid = threadIdx.x;
  const int lane = tid & 31;
  const int wv = tid >> 5;          // 0..7
  const int waveRow = wv & 1;       // 64-row band
  const int waveCol = wv >> 1;      // 32-col band
  const int ln = lane & 15;
  const int lh = lane >> 4;

  __shared__ __align__(16) f16 As[128 * 40];   // [m][k], padded stride 40
  __shared__ __align__(16) f16 Wt[128 * 40];   // [n][k] (transposed W tile)

  v8f acc[4][2];
#pragma unroll
  for (int i = 0; i < 4; ++i)
#pragma unroll
    for (int j = 0; j < 2; ++j) acc[i][j] = (v8f){0.f,0.f,0.f,0.f,0.f,0.f,0.f,0.f};

  for (int k0 = 0; k0 < D_MODEL; k0 += 32) {
    __syncthreads();
    // Stage x tile (128 x 32 f32 -> f16)
#pragma unroll
    for (int i = 0; i < 4; ++i) {
      int idx = tid + 256 * i;
      int row = idx >> 3;
      int c = (idx & 7) * 4;
      const float4 xv = *(const float4*)(x + (size_t)(m0 + row) * D_MODEL + k0 + c);
      f16* d4 = &As[row * 40 + c];
      d4[0] = (f16)xv.x; d4[1] = (f16)xv.y; d4[2] = (f16)xv.z; d4[3] = (f16)xv.w;
    }
    // Stage W tile (32 x 128 f32), store transposed -> Wt[n][k]
#pragma unroll
    for (int i = 0; i < 4; ++i) {
      int idx = tid + 256 * i;
      int r = idx >> 5;                 // k row 0..31
      int c = (idx & 31) * 4;           // n col
      const float4 w4 = *(const float4*)(wp + (size_t)(k0 + r) * D_MODEL + n0 + c);
      Wt[(c + 0) * 40 + r] = (f16)w4.x;
      Wt[(c + 1) * 40 + r] = (f16)w4.y;
      Wt[(c + 2) * 40 + r] = (f16)w4.z;
      Wt[(c + 3) * 40 + r] = (f16)w4.w;
    }
    __syncthreads();

    v16h a[4], b[2];
#pragma unroll
    for (int tr = 0; tr < 4; ++tr)
      a[tr] = load_a_frag(&As[(waveRow * 64 + tr * 16 + ln) * 40], lh);
#pragma unroll
    for (int tc = 0; tc < 2; ++tc)
      b[tc] = load_bt_frag(&Wt[(waveCol * 32 + tc * 16 + ln) * 40], lh);
#pragma unroll
    for (int tr = 0; tr < 4; ++tr)
#pragma unroll
      for (int tc = 0; tc < 2; ++tc)
        acc[tr][tc] = wmma_f16(a[tr], b[tc], acc[tr][tc]);
  }

  // Epilogue: add bias, convert f16, scatter into [b, h, p, d]
#pragma unroll
  for (int tc = 0; tc < 2; ++tc) {
    int n = n0 + waveCol * 32 + tc * 16 + ln;
    float bias = bp[n];
    int hh = n >> 6;
    int d = n & 63;
#pragma unroll
    for (int tr = 0; tr < 4; ++tr) {
#pragma unroll
      for (int r = 0; r < 8; ++r) {
        int m = m0 + waveRow * 64 + tr * 16 + r + 8 * lh;
        int bb = m >> 10;
        int p = m & 1023;
        dst[((size_t)(bb * NUM_HEAD + hh) * SEQ + p) * D_HEAD + d] =
            (f16)(acc[tr][tc][r] + bias);
      }
    }
  }
}

// ---------------------------------------------------------------------------
// Kernel 2: flash attention per (b*h, 64-query tile). 128 threads = 4 waves,
// wave w owns query rows [w*16, w*16+16). KV streamed in 64-row tiles.
// K tile is a pure copy -> async global->LDS path; V tile is transposed in
// flight so it stays on the VGPR path.
// ---------------------------------------------------------------------------
__global__ __launch_bounds__(128)
void mha_attention(const f16* __restrict__ q16, const f16* __restrict__ k16,
                   const f16* __restrict__ v16, f16* __restrict__ o16) {
  const int qt = blockIdx.x;            // 0..15
  const int bh = blockIdx.y;            // 0..127
  const int tid = threadIdx.x;
  const int lane = tid & 31;
  const int wv = tid >> 5;              // 0..3
  const int ln = lane & 15;
  const int lh = lane >> 4;

  const f16* Qb = q16 + (size_t)bh * SEQ * D_HEAD;
  const f16* Kb = k16 + (size_t)bh * SEQ * D_HEAD;
  const f16* Vb = v16 + (size_t)bh * SEQ * D_HEAD;

  __shared__ __align__(16) f16 Kt[64 * 72];       // [kv][d], stride 72
  __shared__ __align__(16) f16 Vt[64 * 72];       // [d][kv] (transposed)
  __shared__ __align__(16) f16 Pl[4][16 * 72];    // per-wave P tile [m][kv]

  // Q fragments live in registers for the whole kernel.
  const f16* qrow = Qb + (size_t)(qt * 64 + wv * 16 + ln) * D_HEAD;
  const v16h aq0 = load_a_frag(qrow + 0, lh);
  const v16h aq1 = load_a_frag(qrow + 32, lh);

  float mrow[8], lrow[8];
  v8f oacc[4];
#pragma unroll
  for (int r = 0; r < 8; ++r) { mrow[r] = -1e30f; lrow[r] = 0.f; }
#pragma unroll
  for (int tc = 0; tc < 4; ++tc) oacc[tc] = (v8f){0.f,0.f,0.f,0.f,0.f,0.f,0.f,0.f};

  for (int t = 0; t < SEQ / 64; ++t) {
    const int kvbase = t * 64;
    __syncthreads();   // previous iteration's Pl/Vt reads complete

    // Stage K tile as-is: Kt[kv][d]
#pragma unroll
    for (int i = 0; i < 4; ++i) {
      int idx = tid + 128 * i;
      int row = idx >> 3;
      int c = (idx & 7) * 8;
#if USE_ASYNC_LDS
      async_copy_b128(Kb + (size_t)(kvbase + row) * D_HEAD + c,
                      &Kt[row * 72 + c]);
#else
      *(v8h*)&Kt[row * 72 + c] =
          *(const v8h*)(Kb + (size_t)(kvbase + row) * D_HEAD + c);
#endif
    }
    // Stage V tile transposed: Vt[d][kv]
#pragma unroll
    for (int i = 0; i < 4; ++i) {
      int idx = tid + 128 * i;
      int kv = idx >> 3;
      int d0 = (idx & 7) * 8;
      v8h val = *(const v8h*)(Vb + (size_t)(kvbase + kv) * D_HEAD + d0);
#pragma unroll
      for (int j = 0; j < 8; ++j) Vt[(d0 + j) * 72 + kv] = val[j];
    }
    if (t + 1 < SEQ / 64) {   // lowers to global_prefetch_b8
      __builtin_prefetch(Kb + (size_t)(kvbase + 64) * D_HEAD, 0, 0);
      __builtin_prefetch(Vb + (size_t)(kvbase + 64) * D_HEAD, 0, 0);
    }
#if USE_ASYNC_LDS
    async_drain();            // ASYNCcnt -> 0 before the barrier
#endif
    __syncthreads();

    // S = Q @ K^T  (raw, scale folded into exp2)
    v8f s[4];
#pragma unroll
    for (int tc = 0; tc < 4; ++tc) {
      v8f c = (v8f){0.f,0.f,0.f,0.f,0.f,0.f,0.f,0.f};
      const f16* krow = &Kt[(tc * 16 + ln) * 72];
      c = wmma_f16(aq0, load_bt_frag(krow + 0, lh), c);
      c = wmma_f16(aq1, load_bt_frag(krow + 32, lh), c);
      s[tc] = c;
    }

    // Online softmax; C layout: vgpr r holds row m = r + 8*lh, col = ln + 16*tc
    float alpha[8];
#pragma unroll
    for (int r = 0; r < 8; ++r) {
      float mx = fmaxf(fmaxf(s[0][r], s[1][r]), fmaxf(s[2][r], s[3][r]));
#pragma unroll
      for (int off = 1; off < 16; off <<= 1)
        mx = fmaxf(mx, __shfl_xor(mx, off, 32));
      float mnew = fmaxf(mrow[r], mx);
      alpha[r] = exp2f((mrow[r] - mnew) * SOFTMAX_C1);
      mrow[r] = mnew;
      float rs = 0.f;
#pragma unroll
      for (int tc = 0; tc < 4; ++tc) {
        float pv = exp2f((s[tc][r] - mnew) * SOFTMAX_C1);
        s[tc][r] = pv;
        rs += pv;
      }
#pragma unroll
      for (int off = 1; off < 16; off <<= 1)
        rs += __shfl_xor(rs, off, 32);
      lrow[r] = lrow[r] * alpha[r] + rs;
    }

    // P (C layout) -> LDS -> A layout
#pragma unroll
    for (int tc = 0; tc < 4; ++tc)
#pragma unroll
      for (int r = 0; r < 8; ++r)
        Pl[wv][(r + 8 * lh) * 72 + tc * 16 + ln] = (f16)s[tc][r];
    __syncthreads();

    const v16h ap0 = load_a_frag(&Pl[wv][ln * 72 + 0], lh);
    const v16h ap1 = load_a_frag(&Pl[wv][ln * 72 + 32], lh);
#pragma unroll
    for (int tc = 0; tc < 4; ++tc) {
      v8f c = oacc[tc];
#pragma unroll
      for (int r = 0; r < 8; ++r) c[r] *= alpha[r];
      const f16* vrow = &Vt[(tc * 16 + ln) * 72];
      c = wmma_f16(ap0, load_bt_frag(vrow + 0, lh), c);
      c = wmma_f16(ap1, load_bt_frag(vrow + 32, lh), c);
      oacc[tc] = c;
    }
  }

  // O /= l, store as f16 into [b, p, h*64+d] for the output GEMM
  const int b = bh >> 4;
  const int h = bh & 15;
#pragma unroll
  for (int tc = 0; tc < 4; ++tc) {
#pragma unroll
    for (int r = 0; r < 8; ++r) {
      int m = qt * 64 + wv * 16 + r + 8 * lh;
      int d = tc * 16 + ln;
      o16[((size_t)b * SEQ + m) * D_MODEL + h * D_HEAD + d] =
          (f16)(oacc[tc][r] / lrow[r]);
    }
  }
}

// ---------------------------------------------------------------------------
// Kernel 3: output projection.  out = O @ w_o + b_o  (fp32 out)
// A tile is a pure f16 copy -> async global->LDS path.
// ---------------------------------------------------------------------------
__global__ __launch_bounds__(256)
void mha_out_proj(const f16* __restrict__ o16, const float* __restrict__ w_o,
                  const float* __restrict__ b_o, float* __restrict__ out) {
  const int m0 = blockIdx.x * 128;
  const int n0 = blockIdx.y * 128;
  const int tid = threadIdx.x;
  const int lane = tid & 31;
  const int wv = tid >> 5;
  const int waveRow = wv & 1;
  const int waveCol = wv >> 1;
  const int ln = lane & 15;
  const int lh = lane >> 4;

  __shared__ __align__(16) f16 As[128 * 40];
  __shared__ __align__(16) f16 Wt[128 * 40];

  v8f acc[4][2];
#pragma unroll
  for (int i = 0; i < 4; ++i)
#pragma unroll
    for (int j = 0; j < 2; ++j) acc[i][j] = (v8f){0.f,0.f,0.f,0.f,0.f,0.f,0.f,0.f};

  for (int k0 = 0; k0 < D_MODEL; k0 += 32) {
    __syncthreads();
#pragma unroll
    for (int i = 0; i < 2; ++i) {        // A tile: 128x32 f16 straight copy
      int idx = tid + 256 * i;
      int row = idx >> 2;
      int c = (idx & 3) * 8;
#if USE_ASYNC_LDS
      async_copy_b128(o16 + (size_t)(m0 + row) * D_MODEL + k0 + c,
                      &As[row * 40 + c]);
#else
      *(v8h*)&As[row * 40 + c] =
          *(const v8h*)(o16 + (size_t)(m0 + row) * D_MODEL + k0 + c);
#endif
    }
#pragma unroll
    for (int i = 0; i < 4; ++i) {        // W tile transposed -> Wt[n][k]
      int idx = tid + 256 * i;
      int r = idx >> 5;
      int c = (idx & 31) * 4;
      const float4 w4 = *(const float4*)(w_o + (size_t)(k0 + r) * D_MODEL + n0 + c);
      Wt[(c + 0) * 40 + r] = (f16)w4.x;
      Wt[(c + 1) * 40 + r] = (f16)w4.y;
      Wt[(c + 2) * 40 + r] = (f16)w4.z;
      Wt[(c + 3) * 40 + r] = (f16)w4.w;
    }
#if USE_ASYNC_LDS
    async_drain();
#endif
    __syncthreads();

    v16h a[4], b[2];
#pragma unroll
    for (int tr = 0; tr < 4; ++tr)
      a[tr] = load_a_frag(&As[(waveRow * 64 + tr * 16 + ln) * 40], lh);
#pragma unroll
    for (int tc = 0; tc < 2; ++tc)
      b[tc] = load_bt_frag(&Wt[(waveCol * 32 + tc * 16 + ln) * 40], lh);
#pragma unroll
    for (int tr = 0; tr < 4; ++tr)
#pragma unroll
      for (int tc = 0; tc < 2; ++tc)
        acc[tr][tc] = wmma_f16(a[tr], b[tc], acc[tr][tc]);
  }

#pragma unroll
  for (int tc = 0; tc < 2; ++tc) {
    int n = n0 + waveCol * 32 + tc * 16 + ln;
    float bias = b_o[n];
#pragma unroll
    for (int tr = 0; tr < 4; ++tr)
#pragma unroll
      for (int r = 0; r < 8; ++r) {
        int m = m0 + waveRow * 64 + tr * 16 + r + 8 * lh;
        out[(size_t)m * D_MODEL + n] = acc[tr][tc][r] + bias;
      }
  }
}

// ---------------------------------------------------------------------------
extern "C" void kernel_launch(void* const* d_in, const int* in_sizes, int n_in,
                              void* d_out, int out_size, void* d_ws, size_t ws_size,
                              hipStream_t stream) {
  const float* x   = (const float*)d_in[0];
  const float* w_q = (const float*)d_in[1];
  const float* b_q = (const float*)d_in[2];
  const float* w_k = (const float*)d_in[3];
  const float* b_k = (const float*)d_in[4];
  const float* w_v = (const float*)d_in[5];
  const float* b_v = (const float*)d_in[6];
  const float* w_o = (const float*)d_in[7];
  const float* b_o = (const float*)d_in[8];

  const size_t elems = (size_t)ROWS * D_MODEL;   // 8,388,608
  f16* qbuf = (f16*)d_ws;
  f16* kbuf = qbuf + elems;
  f16* vbuf = kbuf + elems;
  f16* obuf = vbuf + elems;                      // total 64 MB of f16

  mha_qkv_proj<<<dim3(ROWS / 128, D_MODEL / 128, 3), 256, 0, stream>>>(
      x, w_q, b_q, w_k, b_k, w_v, b_v, qbuf, kbuf, vbuf);

  mha_attention<<<dim3(SEQ / 64, BATCH * NUM_HEAD), 128, 0, stream>>>(
      qbuf, kbuf, vbuf, obuf);

  mha_out_proj<<<dim3(ROWS / 128, D_MODEL / 128), 256, 0, stream>>>(
      obuf, w_o, b_o, (float*)d_out);
}
